// MambaEncoder_22522808500622
// MI455X (gfx1250) — compile-verified, matrix operand order checked by reading, not run on hardware
//
#include <hip/hip_runtime.h>
#include <hip/hip_bf16.h>

typedef __attribute__((ext_vector_type(16))) __bf16 v16bf;
typedef __attribute__((ext_vector_type(8)))  __bf16 v8bf;
typedef __attribute__((ext_vector_type(8)))  float  v8f;
typedef __attribute__((ext_vector_type(4)))  int    v4i;

// ---- model constants ----
constexpr int DM_    = 1024;
constexpr int NL_    = 4;
constexpr int DI_    = 2048;
constexpr int DST_   = 64;          // d_state
constexpr int H_     = 32;
constexpr int P_     = 64;
constexpr int KC_    = 4;           // conv taps
constexpr int CH_    = 64;          // chunk len
constexpr int B_     = 2;
constexpr int S_     = 2048;
constexpr int DCONV_ = DI_ + 2*DST_;        // 2176
constexpr int DIN_   = 2*DI_ + 2*DST_ + H_; // 4256
constexpr int ROWS_  = B_*S_;               // 4096
constexpr int NC_    = S_/CH_;              // 32

// CDNA5 async global->LDS path (ASYNCcnt); fall back to sync staging if the
// toolchain doesn't declare the builtins.
#if defined(__gfx1250__) && __has_builtin(__builtin_amdgcn_global_load_async_to_lds_b128) && \
    __has_builtin(__builtin_amdgcn_s_wait_asynccnt)
#define USE_ASYNC_LDS 1
#else
#define USE_ASYNC_LDS 0
#endif

#define GLOBAL_AS __attribute__((address_space(1)))
#define LDS_AS    __attribute__((address_space(3)))

// builtin expects v4i pointers: global (AS1) src, LDS (AS3) dst
#define TO_GLOBAL_V4I(p) ((GLOBAL_AS v4i*)(v4i*)(void*)(__bf16*)(p))
#define TO_LDS_V4I(p)    ((LDS_AS    v4i*)(v4i*)(void*)(p))

// ---------------- helpers ----------------
__device__ __forceinline__ float block_reduce_sum(float v, float* red) {
  int t = threadIdx.x;
  red[t] = v; __syncthreads();
  for (int s = 128; s > 0; s >>= 1) { if (t < s) red[t] += red[t + s]; __syncthreads(); }
  float r = red[0]; __syncthreads();
  return r;
}
__device__ __forceinline__ float siluf(float x) { return x / (1.0f + expf(-x)); }

// ---------------- embedding gather ----------------
__global__ void embed_kernel(const int* __restrict__ ids, const float* __restrict__ emb,
                             float* __restrict__ out) {
  int row = blockIdx.x;
  int id  = ids[row];
  const float4* src = (const float4*)(emb + (size_t)id * DM_);
  float4* dst = (float4*)(out + (size_t)row * DM_);
  dst[threadIdx.x] = src[threadIdx.x];
}

// ---------------- f32 -> bf16 convert ----------------
__global__ void cvt_bf16_kernel(const float* __restrict__ src, __bf16* __restrict__ dst, int n) {
  int i = blockIdx.x * blockDim.x + threadIdx.x;
  int stride = gridDim.x * blockDim.x;
  for (; i < n; i += stride) dst[i] = (__bf16)src[i];
}

// ---------------- LayerNorm -> bf16 (row = b*s, width DM) ----------------
__global__ void ln_bf16_kernel(const float* __restrict__ x, const float* __restrict__ w,
                               const float* __restrict__ b, __bf16* __restrict__ out) {
  __shared__ float red[256];
  int row = blockIdx.x, t = threadIdx.x;
  float4 v = ((const float4*)(x + (size_t)row * DM_))[t];
  float m = block_reduce_sum(v.x + v.y + v.z + v.w, red) * (1.0f / DM_);
  float d0 = v.x - m, d1 = v.y - m, d2 = v.z - m, d3 = v.w - m;
  float var = block_reduce_sum(d0*d0 + d1*d1 + d2*d2 + d3*d3, red) * (1.0f / DM_);
  float rs = rsqrtf(var + 1e-5f);
  int c = t * 4;
  __bf16* o = out + (size_t)row * DM_;
  o[c+0] = (__bf16)(d0 * rs * w[c+0] + b[c+0]);
  o[c+1] = (__bf16)(d1 * rs * w[c+1] + b[c+1]);
  o[c+2] = (__bf16)(d2 * rs * w[c+2] + b[c+2]);
  o[c+3] = (__bf16)(d3 * rs * w[c+3] + b[c+3]);
}

// ---------------- WMMA bf16 GEMM:  C[M,N] = A[M,K] * W[N,K]^T (+Res) ----------------
// 256 threads = 8 waves. Block tile 128x128, BK=32. Wave tile 64x32 (4x2 frags).
// Double-buffered LDS staging; on CDNA5 the stage uses GLOBAL_LOAD_ASYNC_TO_LDS_B128
// (ASYNCcnt) so tile i+1 streams into LDS while tile i is computed. LDS row stride
// 40 bf16 (80B) keeps every b128 fragment read 16B-aligned and bank-conflict free.
// A-fragment layout per ISA 7.12.2: lane half kb=0/8, elements 0..7 -> K=kb..kb+7,
// elements 8..15 -> K=kb+16..kb+23.
__global__ __launch_bounds__(256) void gemm_bf16_kernel(
    const __bf16* __restrict__ A, const __bf16* __restrict__ W,
    const float* Res, float* C, int M, int N, int K) {
  __shared__ alignas(16) __bf16 As[2][128 * 40];
  __shared__ alignas(16) __bf16 Bs[2][128 * 40];
  union Frag { v16bf v; v8bf h[2]; };

  int t    = threadIdx.x;
  int bm0  = blockIdx.y * 128, bn0 = blockIdx.x * 128;
  int lane = t & 31, wv = t >> 5;
  int wrow = wv >> 2, wcol = wv & 3;        // 2 x 4 wave grid
  int half = lane >> 4, lm = lane & 15, kb = half * 8;

  // this thread's two staging segments (row r, 8-col chunk cs)
  int r0 = t >> 2,          cs0 = (t & 3) * 8;
  int r1 = (t + 256) >> 2,  cs1 = ((t + 256) & 3) * 8;
  int gn0 = bn0 + r0; if (gn0 >= N) gn0 = N - 1;   // clamp OOB weight rows
  int gn1 = bn0 + r1; if (gn1 >= N) gn1 = N - 1;

  v8f acc[4][2];
  #pragma unroll
  for (int i = 0; i < 4; i++)
    #pragma unroll
    for (int j = 0; j < 2; j++)
      #pragma unroll
      for (int e = 0; e < 8; e++) acc[i][j][e] = 0.0f;

#if USE_ASYNC_LDS
  auto stage_async = [&](int k0, int buf) {
    __builtin_amdgcn_global_load_async_to_lds_b128(
        TO_GLOBAL_V4I(A + (size_t)(bm0 + r0) * K + k0 + cs0),
        TO_LDS_V4I(&As[buf][r0 * 40 + cs0]), 0, 0);
    __builtin_amdgcn_global_load_async_to_lds_b128(
        TO_GLOBAL_V4I(A + (size_t)(bm0 + r1) * K + k0 + cs1),
        TO_LDS_V4I(&As[buf][r1 * 40 + cs1]), 0, 0);
    __builtin_amdgcn_global_load_async_to_lds_b128(
        TO_GLOBAL_V4I(W + (size_t)gn0 * K + k0 + cs0),
        TO_LDS_V4I(&Bs[buf][r0 * 40 + cs0]), 0, 0);
    __builtin_amdgcn_global_load_async_to_lds_b128(
        TO_GLOBAL_V4I(W + (size_t)gn1 * K + k0 + cs1),
        TO_LDS_V4I(&Bs[buf][r1 * 40 + cs1]), 0, 0);
  };
  stage_async(0, 0);                        // prologue: tile 0 -> buf 0
#endif

  int cur = 0;
  for (int k0 = 0; k0 < K; k0 += 32) {
#if USE_ASYNC_LDS
    __builtin_amdgcn_s_wait_asynccnt(0);    // this wave's tile-k0 loads landed
    __syncthreads();                        // everyone's landed
    if (k0 + 32 < K) stage_async(k0 + 32, cur ^ 1);  // prior reads of buf^1 done
#else
    __syncthreads();
    {
      const uint4* pa0 = (const uint4*)(A + (size_t)(bm0 + r0) * K + k0 + cs0);
      const uint4* pa1 = (const uint4*)(A + (size_t)(bm0 + r1) * K + k0 + cs1);
      const uint4* pw0 = (const uint4*)(W + (size_t)gn0 * K + k0 + cs0);
      const uint4* pw1 = (const uint4*)(W + (size_t)gn1 * K + k0 + cs1);
      *(uint4*)&As[cur][r0 * 40 + cs0] = *pa0;
      *(uint4*)&As[cur][r1 * 40 + cs1] = *pa1;
      *(uint4*)&Bs[cur][r0 * 40 + cs0] = *pw0;
      *(uint4*)&Bs[cur][r1 * 40 + cs1] = *pw1;
      if (k0 + 32 < K) {
        __builtin_prefetch((const void*)(pa0 + 4), 0, 0);
        __builtin_prefetch((const void*)(pw0 + 4), 0, 0);
      }
    }
    __syncthreads();
#endif

    Frag fa[4], fb[2];
    #pragma unroll
    for (int mi = 0; mi < 4; mi++) {
      int r = wrow * 64 + mi * 16 + lm;
      fa[mi].h[0] = *(const v8bf*)&As[cur][r * 40 + kb];
      fa[mi].h[1] = *(const v8bf*)&As[cur][r * 40 + kb + 16];
    }
    #pragma unroll
    for (int ni = 0; ni < 2; ni++) {
      int r = wcol * 32 + ni * 16 + lm;
      fb[ni].h[0] = *(const v8bf*)&Bs[cur][r * 40 + kb];
      fb[ni].h[1] = *(const v8bf*)&Bs[cur][r * 40 + kb + 16];
    }
    #pragma unroll
    for (int mi = 0; mi < 4; mi++)
      #pragma unroll
      for (int ni = 0; ni < 2; ni++)
        acc[mi][ni] = __builtin_amdgcn_wmma_f32_16x16x32_bf16(
            false, fa[mi].v, false, fb[ni].v, (short)0, acc[mi][ni], false, false);
#if USE_ASYNC_LDS
    cur ^= 1;
#endif
  }

  #pragma unroll
  for (int mi = 0; mi < 4; mi++)
    #pragma unroll
    for (int ni = 0; ni < 2; ni++) {
      int gc = bn0 + wcol * 32 + ni * 16 + lm;
      if (gc < N) {
        #pragma unroll
        for (int r = 0; r < 8; r++) {
          int gr = bm0 + wrow * 64 + mi * 16 + r + 8 * half;  // C/D VGPR layout
          size_t idx = (size_t)gr * N + gc;
          float v = acc[mi][ni][r];
          if (Res) v += Res[idx];
          C[idx] = v;
        }
      }
    }
}

// ---------------- depthwise conv(K=4) + SiLU, softplus(dt) ----------------
__global__ void conv_dt_kernel(const float* __restrict__ zx, const float* __restrict__ cw,
                               const float* __restrict__ cb, const float* __restrict__ dtbias,
                               float* __restrict__ xc, float* __restrict__ dtout) {
  int row = blockIdx.x;             // b*S + s
  int s   = row & (S_ - 1);
  int t   = threadIdx.x;
  for (int c = t; c < DCONV_; c += 256) {
    float acc = cb[c];
    #pragma unroll
    for (int k = 0; k < KC_; k++) {
      int sp = s - 3 + k;
      if (sp >= 0) acc += cw[c * KC_ + k] * zx[(size_t)(row - 3 + k) * DIN_ + DI_ + c];
    }
    xc[(size_t)row * DCONV_ + c] = siluf(acc);
  }
  if (t < H_) {
    float xdt = zx[(size_t)row * DIN_ + DI_ + DCONV_ + t] + dtbias[t];
    dtout[(size_t)row * H_ + t] = (xdt > 20.0f) ? xdt : log1pf(expf(xdt));
  }
}

// ---------------- intra-chunk SSD: y_diag + chunk states ----------------
// one block per (head, chunk, batch); 64x65 fp32 LDS tiles
__global__ __launch_bounds__(256) void chunk_intra_kernel(
    const float* __restrict__ xc, const float* __restrict__ dt,
    const float* __restrict__ alog, const float* __restrict__ Dv,
    float* __restrict__ y, float* __restrict__ acumG,
    float* __restrict__ states, float* __restrict__ dch) {
  __shared__ float S1[64][65];   // B
  __shared__ float S2[64][65];   // C, then X
  __shared__ float S3[64][65];   // G = CB * L * dt
  __shared__ float dts[64], acs[64], wl[64];
  int h = blockIdx.x, c = blockIdx.y, b = blockIdx.z;
  int t = threadIdx.x;
  int srow0 = b * S_ + c * CH_;

  for (int e = t; e < 64 * 64; e += 256) {
    int l = e >> 6, n = e & 63;
    const float* xr = xc + (size_t)(srow0 + l) * DCONV_;
    S1[l][n] = xr[DI_ + n];           // B
    S2[l][n] = xr[DI_ + DST_ + n];    // C
  }
  if (t < 64) dts[t] = dt[(size_t)(srow0 + t) * H_ + h];
  __syncthreads();
  if (t == 0) {
    float A = -expf(alog[h]);
    float run = 0.0f;
    for (int l = 0; l < 64; l++) { run += dts[l] * A; acs[l] = run; }
  }
  __syncthreads();
  if (t < 64) { acumG[(size_t)(srow0 + t) * H_ + h] = acs[t];
                wl[t] = expf(acs[63] - acs[t]) * dts[t]; }
  if (t == 0) dch[((size_t)b * NC_ + c) * H_ + h] = expf(acs[63]);

  for (int e = t; e < 64 * 64; e += 256) {          // G (lower-triangular)
    int l = e >> 6, ss = e & 63;
    float g = 0.0f;
    if (ss <= l) {
      float d = 0.0f;
      for (int n = 0; n < 64; n++) d += S2[l][n] * S1[ss][n];   // C[l]·B[s]
      g = d * expf(acs[l] - acs[ss]) * dts[ss];
    }
    S3[l][ss] = g;
  }
  __syncthreads();
  for (int e = t; e < 64 * 64; e += 256) {          // overwrite S2 with X head slice
    int l = e >> 6, p = e & 63;
    S2[l][p] = xc[(size_t)(srow0 + l) * DCONV_ + h * P_ + p];
  }
  __syncthreads();
  float Dh = Dv[h];
  for (int e = t; e < 64 * 64; e += 256) {          // y_diag + D*x
    int l = e >> 6, p = e & 63;
    float acc = Dh * S2[l][p];
    for (int ss = 0; ss < 64; ss++) acc += S3[l][ss] * S2[ss][p];
    y[(size_t)(srow0 + l) * DI_ + h * P_ + p] = acc;
  }
  size_t sbase = (((size_t)b * NC_ + c) * H_ + h) * (size_t)(P_ * DST_);
  for (int e = t; e < 64 * 64; e += 256) {          // chunk states [p][n]
    int p = e >> 6, n = e & 63;
    float acc = 0.0f;
    for (int l = 0; l < 64; l++) acc += wl[l] * S2[l][p] * S1[l][n];
    states[sbase + e] = acc;
  }
}

// ---------------- inter-chunk scan (sequential over chunks) ----------------
__global__ __launch_bounds__(256) void chunk_scan_kernel(
    const float* __restrict__ xc, const float* __restrict__ acumG,
    const float* __restrict__ states, const float* __restrict__ dch,
    float* __restrict__ y) {
  __shared__ float Sst[64][65];   // running SSM state [p][n]
  __shared__ float Cs[64][65];
  __shared__ float ea[64];
  int h = blockIdx.x, b = blockIdx.y;
  int t = threadIdx.x;
  for (int e = t; e < 64 * 64; e += 256) Sst[e >> 6][e & 63] = 0.0f;
  for (int c = 0; c < NC_; c++) {
    int srow0 = b * S_ + c * CH_;
    __syncthreads();
    for (int e = t; e < 64 * 64; e += 256) {
      int l = e >> 6, n = e & 63;
      Cs[l][n] = xc[(size_t)(srow0 + l) * DCONV_ + DI_ + DST_ + n];
    }
    if (t < 64) ea[t] = expf(acumG[(size_t)(srow0 + t) * H_ + h]);
    __syncthreads();
    for (int e = t; e < 64 * 64; e += 256) {        // y_off uses carry state
      int l = e >> 6, p = e & 63;
      float acc = 0.0f;
      for (int n = 0; n < 64; n++) acc += Cs[l][n] * Sst[p][n];
      y[(size_t)(srow0 + l) * DI_ + h * P_ + p] += ea[l] * acc;
    }
    __syncthreads();
    float dc = dch[((size_t)b * NC_ + c) * H_ + h];
    size_t sbase = (((size_t)b * NC_ + c) * H_ + h) * (size_t)(P_ * DST_);
    for (int e = t; e < 64 * 64; e += 256)
      Sst[e >> 6][e & 63] = dc * Sst[e >> 6][e & 63] + states[sbase + e];
  }
}

// ---------------- SiLU(z) gate + RMSNorm -> bf16 ----------------
__global__ void gate_rms_kernel(const float* __restrict__ zx, const float* __restrict__ y,
                                const float* __restrict__ rmsw, __bf16* __restrict__ out) {
  __shared__ float red[256];
  int row = blockIdx.x, t = threadIdx.x;
  const float* zr = zx + (size_t)row * DIN_;
  const float* yr = y  + (size_t)row * DI_;
  float v[8]; float sq = 0.0f;
  #pragma unroll
  for (int j = 0; j < 8; j++) {
    int d = t + j * 256;
    float val = yr[d] * siluf(zr[d]);
    v[j] = val; sq += val * val;
  }
  float ms = block_reduce_sum(sq, red) * (1.0f / DI_);
  float rs = rsqrtf(ms + 1e-5f);
  __bf16* o = out + (size_t)row * DI_;
  #pragma unroll
  for (int j = 0; j < 8; j++) { int d = t + j * 256; o[d] = (__bf16)(v[j] * rs * rmsw[d]); }
}

// ---------------- mask + final LayerNorm -> f32 out ----------------
__global__ void final_ln_kernel(const float* __restrict__ x, const unsigned char* __restrict__ mask,
                                const float* __restrict__ w, const float* __restrict__ b,
                                float* __restrict__ out) {
  __shared__ float red[256];
  int row = blockIdx.x, t = threadIdx.x;
  float mm = mask[row] ? 0.0f : 1.0f;
  float4 v = ((const float4*)(x + (size_t)row * DM_))[t];
  v.x *= mm; v.y *= mm; v.z *= mm; v.w *= mm;
  float m = block_reduce_sum(v.x + v.y + v.z + v.w, red) * (1.0f / DM_);
  float d0 = v.x - m, d1 = v.y - m, d2 = v.z - m, d3 = v.w - m;
  float var = block_reduce_sum(d0*d0 + d1*d1 + d2*d2 + d3*d3, red) * (1.0f / DM_);
  float rs = rsqrtf(var + 1e-5f);
  float4 w4 = ((const float4*)w)[t];
  float4 b4 = ((const float4*)b)[t];
  float4 o;
  o.x = d0 * rs * w4.x + b4.x;
  o.y = d1 * rs * w4.y + b4.y;
  o.z = d2 * rs * w4.z + b4.z;
  o.w = d3 * rs * w4.w + b4.w;
  ((float4*)(out + (size_t)row * DM_))[t] = o;
}

// ---------------- host ----------------
extern "C" void kernel_launch(void* const* d_in, const int* in_sizes, int n_in,
                              void* d_out, int out_size, void* d_ws, size_t ws_size,
                              hipStream_t stream) {
  (void)in_sizes; (void)n_in; (void)out_size; (void)ws_size;
  const int*           src_ids = (const int*)d_in[0];
  const unsigned char* mask    = (const unsigned char*)d_in[1];
  const float* emb    = (const float*)d_in[2];
  const float* ln_w   = (const float*)d_in[3];
  const float* ln_b   = (const float*)d_in[4];
  const float* in_w   = (const float*)d_in[5];
  const float* conv_w = (const float*)d_in[6];
  const float* conv_b = (const float*)d_in[7];
  const float* dt_b   = (const float*)d_in[8];
  const float* A_log  = (const float*)d_in[9];
  const float* Dv     = (const float*)d_in[10];
  const float* rms_w  = (const float*)d_in[11];
  const float* out_w  = (const float*)d_in[12];
  const float* nw     = (const float*)d_in[13];
  const float* nb     = (const float*)d_in[14];
  float* out = (float*)d_out;

  char* ws = (char*)d_ws;
  size_t off = 0;
  auto alloc = [&](size_t bytes) -> void* {
    void* p = ws + off; off += (bytes + 255) & ~(size_t)255; return p;
  };
  float*  resid = (float*)alloc((size_t)ROWS_ * DM_ * 4);
  __bf16* hbf   = (__bf16*)alloc((size_t)ROWS_ * DM_ * 2);
  __bf16* wbf   = (__bf16*)alloc((size_t)DIN_ * DM_ * 2);
  float*  zx    = (float*)alloc((size_t)ROWS_ * DIN_ * 4);
  float*  xc    = (float*)alloc((size_t)ROWS_ * DCONV_ * 4);
  float*  dtbuf = (float*)alloc((size_t)ROWS_ * H_ * 4);
  float*  acum  = (float*)alloc((size_t)ROWS_ * H_ * 4);
  float*  dchnk = (float*)alloc((size_t)B_ * NC_ * H_ * 4);
  float*  stbuf = (float*)alloc((size_t)B_ * NC_ * H_ * P_ * DST_ * 4);
  float*  ybuf  = (float*)alloc((size_t)ROWS_ * DI_ * 4);
  __bf16* ybf   = (__bf16*)alloc((size_t)ROWS_ * DI_ * 2);

  embed_kernel<<<ROWS_, 256, 0, stream>>>(src_ids, emb, resid);

  for (int i = 0; i < NL_; i++) {
    ln_bf16_kernel<<<ROWS_, 256, 0, stream>>>(resid, ln_w + i * DM_, ln_b + i * DM_, hbf);
    cvt_bf16_kernel<<<2048, 256, 0, stream>>>(in_w + (size_t)i * DIN_ * DM_, wbf, DIN_ * DM_);
    gemm_bf16_kernel<<<dim3((DIN_ + 127) / 128, ROWS_ / 128), 256, 0, stream>>>(
        hbf, wbf, nullptr, zx, ROWS_, DIN_, DM_);
    conv_dt_kernel<<<ROWS_, 256, 0, stream>>>(
        zx, conv_w + (size_t)i * DCONV_ * KC_, conv_b + i * DCONV_, dt_b + i * H_, xc, dtbuf);
    chunk_intra_kernel<<<dim3(H_, NC_, B_), 256, 0, stream>>>(
        xc, dtbuf, A_log + i * H_, Dv + i * H_, ybuf, acum, stbuf, dchnk);
    chunk_scan_kernel<<<dim3(H_, B_), 256, 0, stream>>>(xc, acum, stbuf, dchnk, ybuf);
    gate_rms_kernel<<<ROWS_, 256, 0, stream>>>(zx, ybuf, rms_w + i * DI_, ybf);
    cvt_bf16_kernel<<<2048, 256, 0, stream>>>(out_w + (size_t)i * DM_ * DI_, wbf, DM_ * DI_);
    gemm_bf16_kernel<<<dim3(DM_ / 128, ROWS_ / 128), 256, 0, stream>>>(
        ybf, wbf, resid, resid, ROWS_, DM_, DI_);
  }

  final_ln_kernel<<<ROWS_, 256, 0, stream>>>(resid, mask, nw, nb, out);
}
